// Attention_36833639531186
// MI455X (gfx1250) — matrix operand order repaired and checked by reading, not compiled, not run on hardware
//
#include <hip/hip_runtime.h>

typedef unsigned short u16;
typedef unsigned int   u32;
typedef unsigned long long u64;

typedef __attribute__((ext_vector_type(16))) __bf16 v16bf;
typedef __attribute__((ext_vector_type(8)))  float  v8f;
typedef __attribute__((ext_vector_type(4)))  u32    u32x4;
typedef __attribute__((ext_vector_type(8)))  u32    u32x8;
typedef __attribute__((ext_vector_type(4)))  int    i32x4;
typedef __attribute__((ext_vector_type(8)))  int    i32x8;
typedef __attribute__((ext_vector_type(4)))  float  f32x4;
typedef __attribute__((ext_vector_type(4)))  u16    u16x4;

static __device__ __forceinline__ u16 f2bf(float f) {
    u32 u = __builtin_bit_cast(u32, f);
    u32 r = u + 0x7fffu + ((u >> 16) & 1u);   // round-to-nearest-even
    return (u16)(r >> 16);
}
static __device__ __forceinline__ float bf2f(u16 h) {
    u32 u = ((u32)h) << 16;
    return __builtin_bit_cast(float, u);
}
static __device__ __forceinline__ v16bf frag2(u32x4 a, u32x4 b) {
    u32x8 w = __builtin_shufflevector(a, b, 0, 1, 2, 3, 4, 5, 6, 7);
    return __builtin_bit_cast(v16bf, w);
}
static __device__ __forceinline__ u32x4 ld4(const u16* p) {
    return *(const u32x4*)p;        // 16B-aligned by construction
}

// ---------------------------------------------------------------------------
// TDM: 2D tile load Global -> LDS (D# per CDNA5 ISA ch.8).
// width_elems bf16 per row (must make row bytes = 64), rows rows,
// row stride in elements. pad_enable inserts 16B after every 64B row
// -> LDS row stride = width+8 elements (matches LDSTR=40 for width=32).
// ---------------------------------------------------------------------------
static __device__ __forceinline__ void tdm_load_2d_pad(
    u32 lds_byte_off, const u16* gaddr, u32 width_elems, u32 rows, u32 stride_elems)
{
    u64 ga = (u64)(uintptr_t)gaddr;
    u32x4 g0;
    g0[0] = 1u;                                            // count=1 (valid), user mode
    g0[1] = lds_byte_off;                                  // lds_addr [63:32]
    g0[2] = (u32)ga;                                       // global_addr [95:64]
    g0[3] = (u32)((ga >> 32) & 0x01FFFFFFu) | (2u << 30);  // addr[56:32] | type=2
    i32x8 g1;
    g1[0] = (int)((1u << 16)      // data_size = 2 bytes
                | (1u << 20)      // pad_enable
                | (3u << 22)      // pad_interval: 16 DWORDs (=64B row)
                | (3u << 25));    // pad_amount: 4 DWORDs (=16B = 8 elems)
    g1[1] = (int)(width_elems << 16);   // tensor_dim0[15:0]   (bits 63:48)
    g1[2] = (int)(rows << 16);          // tensor_dim1[15:0]   (bits 95:80)
    g1[3] = (int)(width_elems << 16);   // tile_dim0           (bits 127:112)
    g1[4] = (int)rows;                  // tile_dim1           (bits 143:128)
    g1[5] = (int)stride_elems;          // tensor_dim0_stride  (bits 191:160)
    g1[6] = 0;
    g1[7] = 0;
    i32x4 z4 = {};
#if defined(__clang_major__) && (__clang_major__ >= 23)
    i32x8 z8 = {};
    __builtin_amdgcn_tensor_load_to_lds(g0, g1, z4, z4, z8, 0);
#else
    __builtin_amdgcn_tensor_load_to_lds(g0, g1, z4, z4, 0);
#endif
}

// ---------------------------------------------------------------------------
// K1: fp32 -> bf16 cast, 4 elements/thread
// ---------------------------------------------------------------------------
__global__ void cast_f32_bf16(const float* __restrict__ in, u16* __restrict__ out, int n4) {
    int i = blockIdx.x * blockDim.x + threadIdx.x;
    if (i < n4) {
        f32x4 v = ((const f32x4*)in)[i];
        u16x4 o;
#pragma unroll
        for (int j = 0; j < 4; ++j) o[j] = f2bf(v[j]);
        ((u16x4*)out)[i] = o;
    }
}

// ---------------------------------------------------------------------------
// K2/K5: C[M,N] = A[M,K] @ B[N,K]^T + bias   (A,B bf16, acc f32)
// Block tile 128x128, 8 waves (4M x 2N), wave tile 32x64.
// k-slices staged by the Tensor Data Mover into double-buffered LDS
// (hardware-padded stride 40), fragments via paired ds_load_b128.
// ---------------------------------------------------------------------------
#define LDSTR 40
template <bool F32OUT>
__global__ __launch_bounds__(256) void gemm_bf16_wmma(
    const u16* __restrict__ A, const u16* __restrict__ Bw,
    const float* __restrict__ bias, void* __restrict__ Cout,
    int M, int N, int K)
{
    const int lane = threadIdx.x & 31;
    const int wave = threadIdx.x >> 5;
    const int hi   = lane >> 4;
    const int lj   = lane & 15;
    const int wm   = wave >> 1;          // 0..3
    const int wn   = wave & 1;           // 0..1
    const int m0 = blockIdx.y * 128;
    const int n0 = blockIdx.x * 128;

    __shared__ u16 sA[2][128 * LDSTR];
    __shared__ u16 sB[2][128 * LDSTR];

    const u16* Agb = A  + (size_t)m0 * K;
    const u16* Bgb = Bw + (size_t)n0 * K;
    const u32 offA0 = (u32)(uintptr_t)&sA[0][0];
    const u32 offA1 = (u32)(uintptr_t)&sA[1][0];
    const u32 offB0 = (u32)(uintptr_t)&sB[0][0];
    const u32 offB1 = (u32)(uintptr_t)&sB[1][0];

    v8f acc[2][4] = {};

    // prologue: TDM stages k-slice 0 into buffer 0
    if (wave == 0) {
        tdm_load_2d_pad(offA0, Agb, 32, 128, (u32)K);
        tdm_load_2d_pad(offB0, Bgb, 32, 128, (u32)K);
        __builtin_amdgcn_s_wait_tensorcnt(0);
    }
    __syncthreads();

    const int NIT = K >> 5;
    for (int it = 0; it < NIT; ++it) {
        // issue DMA for next slice while all waves compute this one
        if (it + 1 < NIT && wave == 0) {
            const int kk = (it + 1) << 5;
            const int nb = (it + 1) & 1;
            tdm_load_2d_pad(nb ? offA1 : offA0, Agb + kk, 32, 128, (u32)K);
            tdm_load_2d_pad(nb ? offB1 : offB0, Bgb + kk, 32, 128, (u32)K);
        }

        const u16* a_ = sA[it & 1];
        const u16* b_ = sB[it & 1];
        v16bf af[2];
#pragma unroll
        for (int s = 0; s < 2; ++s) {
            int row = (wm * 32 + s * 16 + lj) * LDSTR;
            af[s] = frag2(ld4(a_ + row + 8 * hi), ld4(a_ + row + 16 + 8 * hi));
        }
#pragma unroll
        for (int t = 0; t < 4; ++t) {
            int nr = (wn * 64 + t * 16 + lj) * LDSTR;
            v16bf bf_ = frag2(ld4(b_ + nr + 8 * hi), ld4(b_ + nr + 16 + 8 * hi));
#pragma unroll
            for (int s = 0; s < 2; ++s)
                acc[s][t] = __builtin_amdgcn_wmma_f32_16x16x32_bf16(
                    false, af[s], false, bf_, (short)0, acc[s][t], false, false);
        }

        if (it + 1 < NIT) {
            if (wave == 0) __builtin_amdgcn_s_wait_tensorcnt(0);
            __syncthreads();
        }
    }

    // branch-free epilogue
#pragma unroll
    for (int s = 0; s < 2; ++s)
#pragma unroll
        for (int t = 0; t < 4; ++t) {
            int n = n0 + wn * 64 + t * 16 + lj;
            float bv = bias[n];
#pragma unroll
            for (int r = 0; r < 8; ++r) {
                int mm = m0 + wm * 32 + s * 16 + r + 8 * hi;
                float val = acc[s][t][r] + bv;
                size_t idx = (size_t)mm * N + n;
                if constexpr (F32OUT) ((float*)Cout)[idx] = val;
                else                  ((u16*)Cout)[idx]   = f2bf(val);
            }
        }
}

// ---------------------------------------------------------------------------
// K3: RoPE + transpose. qkv:[B,T,3,H,D] bf16 -> Q,K:[B,H,T,D], V^T:[B,H,D,T]
// ---------------------------------------------------------------------------
__global__ void rope_kernel(const u16* __restrict__ qkv,
                            u16* __restrict__ Q, u16* __restrict__ Kq,
                            u16* __restrict__ Vt)
{
    int tid = blockIdx.x * blockDim.x + threadIdx.x;  // 2*16*2048*64 total
    int j = tid & 63;
    int t = (tid >> 6) & 2047;
    int h = (tid >> 17) & 15;
    int b = tid >> 21;

    size_t rowbase = ((size_t)(b * 2048 + t)) * 6144;
    int col = h * 128;
    size_t bh = (size_t)(b * 16 + h);
    size_t obase = (bh * 2048 + (size_t)t) * 128;

    float inv = __powf(10000.f, -(float)j * (1.0f / 64.0f));
    float fr  = (float)t * inv;
    float s = __sinf(fr), c = __cosf(fr);

    // q
    float x1 = bf2f(qkv[rowbase + col + 2 * j]);
    float x2 = bf2f(qkv[rowbase + col + 2 * j + 1]);
    Q[obase + j]      = f2bf(x1 * c - x2 * s);
    Q[obase + 64 + j] = f2bf(x1 * s + x2 * c);
    // k
    x1 = bf2f(qkv[rowbase + 2048 + col + 2 * j]);
    x2 = bf2f(qkv[rowbase + 2048 + col + 2 * j + 1]);
    Kq[obase + j]      = f2bf(x1 * c - x2 * s);
    Kq[obase + 64 + j] = f2bf(x1 * s + x2 * c);
    // v: write transposed [B,H,D,T]
    Vt[(bh * 128 + (size_t)(2 * j)) * 2048 + t]     = qkv[rowbase + 4096 + col + 2 * j];
    Vt[(bh * 128 + (size_t)(2 * j + 1)) * 2048 + t] = qkv[rowbase + 4096 + col + 2 * j + 1];
}

// ---------------------------------------------------------------------------
// K4: causal flash attention. 128 thr = 4 waves, 16 q-rows/wave.
// Q,K:[B,H,T,D]; V^T:[B,H,D,T] (PV B-frags are contiguous b128 loads).
// ---------------------------------------------------------------------------
#define PSTR 40
__global__ __launch_bounds__(128) void attn_kernel(
    const u16* __restrict__ Qg, const u16* __restrict__ Kg,
    const u16* __restrict__ Vt, u16* __restrict__ Ob)
{
    const int lane = threadIdx.x & 31;
    const int wave = threadIdx.x >> 5;
    const int hi   = lane >> 4;
    const int lj   = lane & 15;

    const int bh = blockIdx.y;              // b*16 + h
    const int q0 = blockIdx.x * 64 + wave * 16;
    const int b  = bh >> 4;
    const int h  = bh & 15;

    const u16* Qp  = Qg + (size_t)bh * 2048 * 128;
    const u16* Kp  = Kg + (size_t)bh * 2048 * 128;
    const u16* Vtp = Vt + (size_t)bh * 128 * 2048;

    __shared__ u16 ldsP[4][16 * PSTR];
    u16* pp = ldsP[wave];

    v16bf qf[4];
    const int m = q0 + lj;
#pragma unroll
    for (int c = 0; c < 4; ++c) {
        const u16* r = Qp + (size_t)m * 128 + c * 32;
        qf[c] = frag2(ld4(r + 8 * hi), ld4(r + 16 + 8 * hi));
    }

    v8f o[8] = {};
    float mrow[8], lrow[8];
#pragma unroll
    for (int r = 0; r < 8; ++r) { mrow[r] = -3.0e38f; lrow[r] = 0.f; }
    const float scale = 0.08838834764831845f;   // 1/sqrt(128)

    const int jmax = (q0 + 15) >> 5;
    for (int j = 0; j <= jmax; ++j) {
        const int k0 = j * 32;

        v8f s[2];
#pragma unroll
        for (int tt = 0; tt < 2; ++tt) {
            v8f sc = {};
            const int n = k0 + tt * 16 + lj;
            const u16* kr = Kp + (size_t)n * 128;
#pragma unroll
            for (int c = 0; c < 4; ++c) {
                v16bf bf_ = frag2(ld4(kr + c * 32 + 8 * hi),
                                  ld4(kr + c * 32 + 16 + 8 * hi));
                sc = __builtin_amdgcn_wmma_f32_16x16x32_bf16(
                    false, qf[c], false, bf_, (short)0, sc, false, false);
            }
            s[tt] = sc;
        }

#pragma unroll
        for (int r = 0; r < 8; ++r) {
            int qpos = q0 + r + 8 * hi;
            float v0 = s[0][r] * scale;
            float v1 = s[1][r] * scale;
            if (k0 + lj > qpos)      v0 = -3.0e38f;
            if (k0 + 16 + lj > qpos) v1 = -3.0e38f;

            float cand = fmaxf(v0, v1);
#pragma unroll
            for (int off = 1; off < 16; off <<= 1)
                cand = fmaxf(cand, __shfl_xor(cand, off));
            float newm  = fmaxf(mrow[r], cand);
            float alpha = __expf(mrow[r] - newm);
            mrow[r] = newm;

            v0 = __expf(v0 - newm);
            v1 = __expf(v1 - newm);
            s[0][r] = v0;
            s[1][r] = v1;

            float rs = v0 + v1;
#pragma unroll
            for (int off = 1; off < 16; off <<= 1)
                rs += __shfl_xor(rs, off);
            lrow[r] = lrow[r] * alpha + rs;

#pragma unroll
            for (int t = 0; t < 8; ++t) o[t][r] *= alpha;
        }

        // P (C layout) -> LDS bf16, re-read as A fragment
#pragma unroll
        for (int tt = 0; tt < 2; ++tt)
#pragma unroll
            for (int r = 0; r < 8; ++r)
                pp[(r + 8 * hi) * PSTR + tt * 16 + lj] = f2bf(s[tt][r]);

        v16bf pa = frag2(ld4(pp + lj * PSTR + 8 * hi),
                         ld4(pp + lj * PSTR + 16 + 8 * hi));

#pragma unroll
        for (int t = 0; t < 8; ++t) {
            const u16* vr = Vtp + (size_t)(t * 16 + lj) * 2048 + k0 + 8 * hi;
            v16bf bf_ = frag2(ld4(vr), ld4(vr + 16));
            o[t] = __builtin_amdgcn_wmma_f32_16x16x32_bf16(
                false, pa, false, bf_, (short)0, o[t], false, false);
        }
    }

#pragma unroll
    for (int t = 0; t < 8; ++t) {
        int d = t * 16 + lj;
#pragma unroll
        for (int r = 0; r < 8; ++r) {
            int trow = q0 + r + 8 * hi;
            float val = o[t][r] * (1.0f / lrow[r]);
            size_t idx = ((size_t)(b * 2048 + trow)) * 2048 + (size_t)(h * 128 + d);
            Ob[idx] = f2bf(val);
        }
    }
}

// ---------------------------------------------------------------------------
// Launch
// ---------------------------------------------------------------------------
extern "C" void kernel_launch(void* const* d_in, const int* in_sizes, int n_in,
                              void* d_out, int out_size, void* d_ws, size_t ws_size,
                              hipStream_t stream) {
    const float* x    = (const float*)d_in[0];   // [2,2048,2048]
    const float* Wqkv = (const float*)d_in[1];   // [6144,2048]
    const float* bqkv = (const float*)d_in[2];   // [6144]
    const float* Wout = (const float*)d_in[3];   // [2048,2048]
    const float* bout = (const float*)d_in[4];   // [2048]
    float* out = (float*)d_out;                  // [2,2048,2048] f32

    char* ws = (char*)d_ws;
    u16* xb    = (u16*)(ws + 0);            //  16.8 MB  x bf16
    u16* wqkvb = (u16*)(ws + 16777216);     //  25.2 MB  W_qkv bf16
    u16* wob   = (u16*)(ws + 41943040);     //   8.4 MB  W_out bf16
    u16* qkvb  = (u16*)(ws + 50331648);     //  50.3 MB  qkv bf16
    u16* Qb    = (u16*)(ws + 100663296);    //  16.8 MB  Q [B,H,T,D]
    u16* Kb    = (u16*)(ws + 117440512);    //  16.8 MB  K [B,H,T,D]
    u16* Vtb   = (u16*)(ws + 134217728);    //  16.8 MB  V^T [B,H,D,T]
    u16* Ob    = (u16*)(ws + 150994944);    //  16.8 MB  attn out [B,T,C]

    cast_f32_bf16<<<8388608 / 4 / 256, 256, 0, stream>>>(x, xb, 8388608 / 4);
    cast_f32_bf16<<<12582912 / 4 / 256, 256, 0, stream>>>(Wqkv, wqkvb, 12582912 / 4);
    cast_f32_bf16<<<4194304 / 4 / 256, 256, 0, stream>>>(Wout, wob, 4194304 / 4);

    // qkv = x @ Wqkv^T + b   (M=4096, N=6144, K=2048) -> bf16
    gemm_bf16_wmma<false><<<dim3(6144 / 128, 4096 / 128), 256, 0, stream>>>(
        xb, wqkvb, bqkv, qkvb, 4096, 6144, 2048);

    rope_kernel<<<(2 * 16 * 2048 * 64) / 256, 256, 0, stream>>>(qkvb, Qb, Kb, Vtb);

    attn_kernel<<<dim3(2048 / 64, 2 * 16), 128, 0, stream>>>(Qb, Kb, Vtb, Ob);

    // out = O @ Wout^T + b_out  (M=4096, N=2048, K=2048) -> f32
    gemm_bf16_wmma<true><<<dim3(2048 / 128, 4096 / 128), 256, 0, stream>>>(
        Ob, wob, bout, out, 4096, 2048, 2048);
}
